// Conv4d_41042707480608
// MI455X (gfx1250) — matrix operand order repaired and checked by reading, not compiled, not run on hardware
//
#include <hip/hip_runtime.h>

// ---------------- problem constants ----------------
#define N_   2
#define IC   16
#define OC   32
// x strides in elements, shape (2,16,20,20,20,20)
#define XC   160000   // 20^4
#define XI   8000     // 20^3
#define XJ   400
#define XK   20
// out shape (2,32,18,18,18,18)
#define OD   18
#define OSP  104976   // 18^4
#define OI   5832     // 18^3
#define OJ   324
#define OK   18
#define TAPS 81
#define KTOT 1296     // 16 * 81
#define KB   41       // K-blocks of 32 (last half-block zero-padded in B)

// LDS slab: spatial rows [a(3)][b(3)][dk(6)][dl(20)] = 1080 rows, channel-innermost,
// channel stride padded 16 -> 18 elements (conflict-free + keeps 8B alignment).
#define SROWS 1080
#define CPAD  18

typedef __attribute__((ext_vector_type(16))) __bf16 v16bf;
typedef __attribute__((ext_vector_type(8)))  float  v8f;
typedef __attribute__((ext_vector_type(2)))  float  f2;

__device__ __forceinline__ int tapDigits(int t, int& a, int& b, int& c, int& d) {
    a = t / 27; int r = t - a * 27;
    b = r / 9;  r -= b * 9;
    c = r / 3;  d = r - c * 3;
    return 0;
}

// element offset of the start of the row holding (tap, kkLoc, l0) in the slab
__device__ __forceinline__ int slabRowBase(int tap, int kkLoc, int l0) {
    int a, b, c, d;
    tapDigits(tap, a, b, c, d);
    return (((a * 3 + b) * 6 + kkLoc + c) * 20 + l0 + d) * CPAD;
}

// ---------------------------------------------------------------------------
// Repack w (f32 [16][32][81]) into bf16 hi/lo B-fragments in d_ws.
// Fragment f = ((kb*2 + nhalf)*2 + term), 32 lanes x 16 bf16 per fragment.
// B layout (32x16 KxN, 16-bit): lane = {Khalf, N}, element e -> K = Khalf*8 + e (e<8), 16 + Khalf*8 + (e-8) (e>=8).
// k = kb*32 + K ; k = tap*16 + c ; zero-pad k >= 1296.
// ---------------------------------------------------------------------------
__global__ void repack_w_kernel(const float* __restrict__ w, __bf16* __restrict__ wsB) {
    int t = blockIdx.x * 256 + threadIdx.x;
    const int total = KB * 2 * 2 * 32 * 16;   // 83,968 bf16 (164 KB)
    if (t >= total) return;
    int e    = t & 15;
    int lane = (t >> 4) & 31;
    int f    = t >> 9;
    int term = f & 1;
    int nh   = (f >> 1) & 1;
    int kb   = f >> 2;
    int K = (e < 8) ? ((lane >> 4) * 8 + e) : (16 + (lane >> 4) * 8 + (e - 8));
    int k = kb * 32 + K;
    float val = 0.0f;
    if (k < KTOT) {
        int tap = k >> 4;
        int c   = k & 15;
        int o   = nh * 16 + (lane & 15);
        val = w[(c * OC + o) * TAPS + tap];
    }
    __bf16 hv = (__bf16)val;
    wsB[t] = (term == 0) ? hv : (__bf16)(val - (float)hv);
}

// ---------------------------------------------------------------------------
// Main kernel: block = 8 waves sharing one LDS x-slab; each wave32 computes a
// 16-position x 32-outchannel tile via v_wmma_f32_16x16x32_bf16 (hi/lo split).
// ---------------------------------------------------------------------------
__global__ void __launch_bounds__(256) conv4d_wmma_kernel(
    const float* __restrict__ x,
    const __bf16* __restrict__ wsB,
    float* __restrict__ out) {

    __shared__ float slab[SROWS * CPAD];   // 77,760 B

    const int tid  = threadIdx.x;
    const int lane = tid & 31;
    const int wv   = tid >> 5;             // wave in block: 0..7

    // block decomposition: (n, i, j, kk-group)
    int b = blockIdx.x;
    const int g = b % 5;  b /= 5;
    const int j = b % OD; b /= OD;
    const int i = b % OD;
    const int n = b / OD;
    const int kkBase = (g < 4) ? 4 * g : 14;   // group 4 overlaps group 3 (identical recompute)

    // ---- cooperative stage: x slab -> LDS, channel-innermost (padded) ----
    const int gbase = n * (IC * XC) + i * XI + j * XJ + kkBase * XK;
    for (int s = tid; s < SROWS * IC; s += 256) {
        int row = s % SROWS;               // consecutive tid -> consecutive dl: coalesced
        int c   = s / SROWS;
        int di = row / 360, r2 = row - di * 360;
        int dj = r2 / 120,  r3 = r2 - dj * 120;
        int dk = r3 / 20,   dl = r3 - dk * 20;
        slab[row * CPAD + c] = x[gbase + c * XC + di * XI + dj * XJ + dk * XK + dl];
    }
    __syncthreads();

    const int kkLoc = wv >> 1;             // 0..3
    const int l0    = (wv & 1) << 1;       // 0 or 2 (18 = 16 + 2)
    const int sel   = lane >> 4;           // K-half selector per ISA A-layout
    const int m     = lane & 15;           // M = position within the l tile
    const int laneOff = m * CPAD + sel * 8;

    v8f acc0 = {};   // out-channels  0..15
    v8f acc1 = {};   // out-channels 16..31
    const v16bf* __restrict__ Bfrag = (const v16bf*)wsB;

    for (int kb = 0; kb < KB; ++kb) {
        const int tap0 = kb * 2;
        const int tap1 = tap0 + 1;
        const int base0 = slabRowBase(tap0, kkLoc, l0);                            // wave-uniform
        const int base1 = (tap1 < TAPS) ? slabRowBase(tap1, kkLoc, l0) : base0;    // pad: B is zero
        const float* p0 = &slab[base0 + laneOff];
        const float* p1 = &slab[base1 + laneOff];

        // ---- A hi/lo fragments: 8 x ds_load_b64 (adjacent-channel K-pairs) ----
        v16bf ah, al;
#pragma unroll
        for (int v = 0; v < 8; ++v) {
            const float* p = (v < 4) ? p0 : p1;
            const int cw = (v < 4) ? (2 * v) : (2 * (v - 4));
            f2 xv = *(const f2*)(p + cw);              // 8B-aligned -> ds_load_b64
#pragma unroll
            for (int q = 0; q < 2; ++q) {
                __bf16 h = (__bf16)xv[q];
                __bf16 l = (__bf16)(xv[q] - (float)h);
                ah[2 * v + q] = h;
                al[2 * v + q] = l;
            }
        }

        // ---- B fragments (pre-repacked, contiguous 32B per lane, L0-resident) ----
        const int fb = kb * 4;
        v16bf bh0 = Bfrag[(fb + 0) * 32 + lane];
        v16bf bl0 = Bfrag[(fb + 1) * 32 + lane];
        v16bf bh1 = Bfrag[(fb + 2) * 32 + lane];
        v16bf bl1 = Bfrag[(fb + 3) * 32 + lane];
        if (kb + 1 < KB)
            __builtin_prefetch(&Bfrag[(fb + 4) * 32 + lane], 0, 1);   // global_prefetch_b8

        // ---- 6 WMMAs: (Ah+Al)x(Bh+Bl) dropping lo*lo, f32 accumulate ----
        acc0 = __builtin_amdgcn_wmma_f32_16x16x32_bf16(false, ah, false, bh0, (short)0, acc0, false, false);
        acc0 = __builtin_amdgcn_wmma_f32_16x16x32_bf16(false, al, false, bh0, (short)0, acc0, false, false);
        acc0 = __builtin_amdgcn_wmma_f32_16x16x32_bf16(false, ah, false, bl0, (short)0, acc0, false, false);
        acc1 = __builtin_amdgcn_wmma_f32_16x16x32_bf16(false, ah, false, bh1, (short)0, acc1, false, false);
        acc1 = __builtin_amdgcn_wmma_f32_16x16x32_bf16(false, al, false, bh1, (short)0, acc1, false, false);
        acc1 = __builtin_amdgcn_wmma_f32_16x16x32_bf16(false, ah, false, bl1, (short)0, acc1, false, false);
    }

    // ---- store: C/D layout -> VGPR r, lane: M = r + 8*sel, N = lane&15 ----
    const int kk    = kkBase + kkLoc;
    const int obase = n * (OC * OSP) + i * OI + j * OJ + kk * OK + l0;
    const int ncol  = lane & 15;
#pragma unroll
    for (int r = 0; r < 8; ++r) {
        const int M = r + 8 * sel;
        out[obase + ncol * OSP + M]        = acc0[r];
        out[obase + (16 + ncol) * OSP + M] = acc1[r];
    }
}

extern "C" void kernel_launch(void* const* d_in, const int* in_sizes, int n_in,
                              void* d_out, int out_size, void* d_ws, size_t ws_size,
                              hipStream_t stream) {
    (void)in_sizes; (void)n_in; (void)out_size; (void)ws_size;
    const float* x = (const float*)d_in[0];
    const float* w = (const float*)d_in[1];
    float* out = (float*)d_out;
    __bf16* wsB = (__bf16*)d_ws;     // needs 164 KB of scratch

    const int totalB = KB * 2 * 2 * 32 * 16;
    repack_w_kernel<<<(totalB + 255) / 256, 256, 0, stream>>>(w, wsB);

    const int blocks = N_ * OD * OD * 5;   // 3240 blocks x 8 waves
    conv4d_wmma_kernel<<<blocks, 256, 0, stream>>>(x, wsB, out);
}
